// Long_Run_38878043963796
// MI455X (gfx1250) — matrix operand order repaired and checked by reading, not compiled, not run on hardware
//
#include <hip/hip_runtime.h>

#define B_N   32768
#define D_RGB 512
#define D_SPD 128
#define D_IN  640
#define H_W   256
#define N_C   7
#define MAXG  (B_N/128 + N_C)

typedef __bf16    v16bf __attribute__((ext_vector_type(16)));
typedef float     v8f   __attribute__((ext_vector_type(8)));
typedef float     f32x4 __attribute__((ext_vector_type(4)));
typedef unsigned  u32x4 __attribute__((ext_vector_type(4)));

union Frag16 { unsigned u[8]; u32x4 q[2]; v16bf v; };
union ARegs  { f32x4 f[4]; u32x4 u[2]; };

__device__ __forceinline__ int imin(int a, int b) { return a < b ? a : b; }

__device__ __forceinline__ unsigned short f2bf(float f) {
  union { float f; unsigned u; } v; v.f = f;
  unsigned r = v.u + 0x7FFFu + ((v.u >> 16) & 1u);   // round-to-nearest-even
  return (unsigned short)(r >> 16);
}
__device__ __forceinline__ float bf2f(unsigned short h) {
  union { unsigned u; float f; } v; v.u = ((unsigned)h) << 16; return v.f;
}
__device__ __forceinline__ float sigm(float x) { return 1.f / (1.f + __expf(-x)); }

__device__ __forceinline__ v8f wmma_bf16(v16bf a, v16bf b, v8f c) {
  return __builtin_amdgcn_wmma_f32_16x16x32_bf16(false, a, false, b, (short)0, c, false, false);
}

// ---- B-tile staging, split into load (global->VGPR) and store (VGPR->LDS swizzled) ----
// LDS B layout: frag(nt,lane) = 16 contiguous bf16 at ((nt*32+lane)*16);
// element (v,h) = B[k][n], k = 2v + h + 16*(lane>>4), n = nt*16 + (lane&15).
template <int NCOL>
__device__ __forceinline__ void loadB(const unsigned short* __restrict__ gB, int tid, u32x4* r) {
  const u32x4* g = (const u32x4*)gB;
#pragma unroll
  for (int it = 0; it < NCOL / 64; ++it)
    r[it] = g[it * 256 + tid];                 // 16B per thread per chunk, fully coalesced
}

template <int NCOL>
__device__ __forceinline__ void storeB(unsigned short* sB, int tid, const u32x4* r) {
#pragma unroll
  for (int it = 0; it < NCOL / 64; ++it) {
    int e0 = (it * 256 + tid) * 4;             // first bf16-pair index of this chunk
    int k  = e0 / (NCOL / 2);
    int n0 = (e0 % (NCOL / 2)) * 2;            // multiple of 8 -> stays in one nt group
    int lh = k >> 4, rem = k & 15, vv = rem >> 1, hh = rem & 1;
    int nt = n0 >> 4;
    int lane0 = (n0 & 15) + (lh << 4);
    int base = ((nt * 32 + lane0) << 4) + (vv << 1) + hh;
    unsigned v0 = r[it].x, v1 = r[it].y, v2 = r[it].z, v3 = r[it].w;
    sB[base + 0 * 16] = (unsigned short)v0;  sB[base + 1 * 16] = (unsigned short)(v0 >> 16);
    sB[base + 2 * 16] = (unsigned short)v1;  sB[base + 3 * 16] = (unsigned short)(v1 >> 16);
    sB[base + 4 * 16] = (unsigned short)v2;  sB[base + 5 * 16] = (unsigned short)(v2 >> 16);
    sB[base + 6 * 16] = (unsigned short)v3;  sB[base + 7 * 16] = (unsigned short)(v3 >> 16);
  }
}

// ---- A fragment from LDS row-major (padded), contiguous k = 16*(lane>>4) + s, same order as B ----
__device__ __forceinline__ v16bf load_A(const unsigned* sA32, int rowBaseDw, int kOff, int lane) {
  Frag16 f;
  int lh = lane >> 4;
#pragma unroll
  for (int vv = 0; vv < 8; ++vv) {
    int p = vv << 1;
    int K = p + (lh << 4);  // even
    f.u[vv] = sA32[rowBaseDw + ((kOff + K) >> 1)];
  }
  return f.v;
}

__device__ __forceinline__ v16bf load_B(const unsigned short* sB, int nt, int lane) {
  Frag16 f;
  const u32x4* p = (const u32x4*)(sB + ((nt * 32 + lane) << 4));
  f.q[0] = p[0]; f.q[1] = p[1];
  return f.v;
}

__device__ __forceinline__ v8f v8f_zero() {
  v8f z;
#pragma unroll
  for (int i = 0; i < 8; ++i) z[i] = 0.f;
  return z;
}

// ---- layer-1 A staging (gathered row, rgb f32 region / sfeat bf16 region) ----
__device__ __forceinline__ void loadA1(int kt, const float* rgbRow,
                                       const unsigned short* sfRow, int scol0, ARegs& ar) {
  if (kt < 16) {
    const f32x4* s = (const f32x4*)(rgbRow + kt * 32 + scol0);
#pragma unroll
    for (int q = 0; q < 4; ++q) ar.f[q] = s[q];
  } else {
    const u32x4* s = (const u32x4*)(sfRow + (kt - 16) * 32 + scol0);
    ar.u[0] = s[0]; ar.u[1] = s[1];
  }
}
__device__ __forceinline__ void storeA1(int kt, const ARegs& ar, unsigned* dst32) {
  if (kt < 16) {
#pragma unroll
    for (int q = 0; q < 4; ++q) {
      f32x4 v4 = ar.f[q];
      dst32[q * 2 + 0] = (unsigned)f2bf(v4.x) | ((unsigned)f2bf(v4.y) << 16);
      dst32[q * 2 + 1] = (unsigned)f2bf(v4.z) | ((unsigned)f2bf(v4.w) << 16);
    }
  } else {
    dst32[0] = ar.u[0].x; dst32[1] = ar.u[0].y; dst32[2] = ar.u[0].z; dst32[3] = ar.u[0].w;
    dst32[4] = ar.u[1].x; dst32[5] = ar.u[1].y; dst32[6] = ar.u[1].z; dst32[7] = ar.u[1].w;
  }
}

// ---------------- small helper kernels ----------------

__global__ void k_init(int* hdr) { if (threadIdx.x < 32) hdr[threadIdx.x] = 0; }

__global__ void k_cvt(const float* __restrict__ src, unsigned short* __restrict__ dst, int n) {
  int i = blockIdx.x * 256 + threadIdx.x;
  if (i < n) dst[i] = f2bf(src[i]);
}

__global__ void k_spd1(const float* __restrict__ spd, const float* __restrict__ W1,
                       const float* __restrict__ b1, unsigned short* __restrict__ s1) {
  int i = blockIdx.x * 256 + threadIdx.x;
  int b = i >> 7, j = i & 127;
  s1[i] = f2bf(tanhf(spd[b] * W1[j] + b1[j]));
}

__global__ void k_hist(const int* __restrict__ cmd, int* __restrict__ hdr) {
  __shared__ int hc[N_C];
  int tid = threadIdx.x;
  if (tid < N_C) hc[tid] = 0;
  __syncthreads();
  int b = blockIdx.x * 256 + tid;
  atomicAdd(&hc[cmd[b]], 1);
  __syncthreads();
  if (tid < N_C) atomicAdd(&hdr[tid], hc[tid]);
}

__global__ void k_scan(int* hdr) {
  if (threadIdx.x == 0 && blockIdx.x == 0) {
    int o = 0, g = 0;
    for (int c = 0; c < N_C; ++c) {
      hdr[16 + c] = o;
      hdr[24 + c] = g;
      hdr[8 + c]  = 0;
      o += hdr[c];
      g += (hdr[c] + 127) >> 7;
    }
    hdr[16 + N_C] = o;
    hdr[24 + N_C] = g;
  }
}

__global__ void k_scatter(const int* __restrict__ cmd, int* __restrict__ hdr,
                          int* __restrict__ rowidx) {
  int b = blockIdx.x * 256 + threadIdx.x;
  int c = cmd[b];
  int pos = hdr[16 + c] + atomicAdd(&hdr[8 + c], 1);
  rowidx[pos] = b;
}

// ---------------- speed MLP GEMM: out = tanh(in[B,128] x W[128,128] + b) ----------------

__global__ __launch_bounds__(256)
void k_spd_gemm(const unsigned short* __restrict__ sin, const unsigned short* __restrict__ Wbf,
                const float* __restrict__ bias, unsigned short* __restrict__ sout) {
  __shared__ __align__(16) unsigned short sB[2][8 * 32 * 16];
  __shared__ __align__(16) unsigned short sAst[2][128 * 36];

  int tid = threadIdx.x, wave = tid >> 5, lane = tid & 31;
  int m = lane & 15, lh = lane >> 4;
  int rowBlk = blockIdx.x * 128;

  v8f acc[8];
  v8f z = v8f_zero();
#pragma unroll
  for (int i = 0; i < 8; ++i) acc[i] = z;

  int sr = tid >> 1;
  int scol0 = (tid & 1) * 16;
  int dbase = sr * 18 + (scol0 >> 1);
  const unsigned short* aRow = sin + (size_t)(rowBlk + sr) * D_SPD + scol0;

  u32x4 br[2];
  u32x4 arr[2];
  loadB<D_SPD>(Wbf, tid, br);
  { const u32x4* s = (const u32x4*)aRow; arr[0] = s[0]; arr[1] = s[1]; }

  for (int kt = 0; kt < 4; ++kt) {
    unsigned short* sBb = sB[kt & 1];
    unsigned* dA = (unsigned*)sAst[kt & 1] + dbase;
    storeB<D_SPD>(sBb, tid, br);
    dA[0] = arr[0].x; dA[1] = arr[0].y; dA[2] = arr[0].z; dA[3] = arr[0].w;
    dA[4] = arr[1].x; dA[5] = arr[1].y; dA[6] = arr[1].z; dA[7] = arr[1].w;
    __syncthreads();
    if (kt < 3) {
      loadB<D_SPD>(Wbf + (kt + 1) * 32 * D_SPD, tid, br);
      const u32x4* s = (const u32x4*)(aRow + (kt + 1) * 32);
      arr[0] = s[0]; arr[1] = s[1];
    }
    v16bf a = load_A((const unsigned*)sAst[kt & 1], (wave * 16 + m) * 18, 0, lane);
#pragma unroll
    for (int nt = 0; nt < 8; ++nt) {
      v16bf b = load_B(sBb, nt, lane);
      acc[nt] = wmma_bf16(a, b, acc[nt]);
    }
    __syncthreads();
  }
#pragma unroll
  for (int nt = 0; nt < 8; ++nt) {
    float bia = bias[nt * 16 + m];
#pragma unroll
    for (int r = 0; r < 8; ++r) {
      int row = rowBlk + wave * 16 + r + lh * 8;
      sout[(size_t)row * D_SPD + nt * 16 + m] = f2bf(tanhf(acc[nt][r] + bia));
    }
  }
}

// ---------------- main fused branch MLP (selected branch only, bucketed rows) ----------------

__global__ __launch_bounds__(256)
void k_branch(const float* __restrict__ rgb, const unsigned short* __restrict__ sfeat,
              const int* __restrict__ hdr, const int* __restrict__ rowidx,
              const unsigned short* __restrict__ W1bf, const unsigned short* __restrict__ W2bf,
              const unsigned short* __restrict__ W3bf,
              const float* __restrict__ b1, const float* __restrict__ b2,
              const float* __restrict__ b3,
              const float* __restrict__ hWb, const float* __restrict__ hbb,
              const float* __restrict__ hWs, const float* __restrict__ hbs,
              const float* __restrict__ hWt, const float* __restrict__ hbt,
              float* __restrict__ out) {
  __shared__ __align__(16) unsigned short sB[2][16 * 32 * 16];  // 2 x 16 KB, double-buffered
  __shared__ __align__(16) unsigned short sAst[2][128 * 36];    // 2 x 9 KB (layer-1 staging)
  __shared__ __align__(16) unsigned short sH[128 * 264];        // 66 KB activations (padded)

  int g = blockIdx.x;
  const int* gs   = hdr + 24;
  const int* offs = hdr + 16;
  if (g >= gs[N_C]) return;
  int c = 0;
  while (c < N_C - 1 && g >= gs[c + 1]) ++c;
  int bucketBase = offs[c];
  int cnt        = offs[c + 1] - offs[c];
  int glocal     = g - gs[c];

  int tid = threadIdx.x, wave = tid >> 5, lane = tid & 31;
  int m = lane & 15, lh = lane >> 4;

  const unsigned short* Wc1 = W1bf + (size_t)c * D_IN * H_W;
  const unsigned short* Wc2 = W2bf + (size_t)c * H_W * H_W;
  const unsigned short* Wc3 = W3bf + (size_t)c * H_W * H_W;

  int sr = tid >> 1;
  int scol0 = (tid & 1) * 16;
  int srow = rowidx[bucketBase + imin(glocal * 128 + sr, cnt - 1)];
  const float* rgbRow = rgb + (size_t)srow * D_RGB;
  const unsigned short* sfRow = sfeat + (size_t)srow * D_SPD;
  int dbase = sr * 18 + (scol0 >> 1);

  v8f acc[16];
  v8f z = v8f_zero();

  // ---------------- layer 1: x[128,640] x W1[640,256] ----------------
#pragma unroll
  for (int i = 0; i < 16; ++i) acc[i] = z;
  {
    u32x4 br4[4];
    ARegs ar;
    loadB<H_W>(Wc1, tid, br4);
    loadA1(0, rgbRow, sfRow, scol0, ar);
    for (int kt = 0; kt < 20; ++kt) {
      unsigned short* sBb = sB[kt & 1];
      storeB<H_W>(sBb, tid, br4);
      storeA1(kt, ar, (unsigned*)sAst[kt & 1] + dbase);
      __syncthreads();
      if (kt < 19) {
        loadB<H_W>(Wc1 + (kt + 1) * 32 * H_W, tid, br4);
        loadA1(kt + 1, rgbRow, sfRow, scol0, ar);
      }
      v16bf a = load_A((const unsigned*)sAst[kt & 1], (wave * 16 + m) * 18, 0, lane);
#pragma unroll
      for (int nt = 0; nt < 16; ++nt) {
        v16bf b = load_B(sBb, nt, lane);
        acc[nt] = wmma_bf16(a, b, acc[nt]);
      }
      __syncthreads();
    }
  }
  // epilogue: each wave writes only its own 16 rows of sH (no cross-wave hazard)
#pragma unroll
  for (int nt = 0; nt < 16; ++nt) {
    float bia = b1[c * H_W + nt * 16 + m];
#pragma unroll
    for (int r = 0; r < 8; ++r)
      sH[(wave * 16 + r + lh * 8) * 264 + nt * 16 + m] = f2bf(sigm(acc[nt][r] + bia));
  }

  // ---------------- layers 2 & 3: h[128,256] x W[256,256] ----------------
  for (int layer = 0; layer < 2; ++layer) {
    const unsigned short* Wc = (layer == 0) ? Wc2 : Wc3;
    const float* bb = (layer == 0) ? b2 : b3;
#pragma unroll
    for (int i = 0; i < 16; ++i) acc[i] = z;
    u32x4 br4[4];
    loadB<H_W>(Wc, tid, br4);
    for (int kt = 0; kt < 8; ++kt) {
      unsigned short* sBb = sB[kt & 1];
      storeB<H_W>(sBb, tid, br4);
      __syncthreads();
      if (kt < 7) loadB<H_W>(Wc + (kt + 1) * 32 * H_W, tid, br4);
      v16bf a = load_A((const unsigned*)sH, (wave * 16 + m) * 132, kt * 32, lane);
#pragma unroll
      for (int nt = 0; nt < 16; ++nt) {
        v16bf b = load_B(sBb, nt, lane);
        acc[nt] = wmma_bf16(a, b, acc[nt]);
      }
      __syncthreads();
    }
#pragma unroll
    for (int nt = 0; nt < 16; ++nt) {
      float bia = bb[c * H_W + nt * 16 + m];
#pragma unroll
      for (int r = 0; r < 8; ++r)
        sH[(wave * 16 + r + lh * 8) * 264 + nt * 16 + m] = f2bf(sigm(acc[nt][r] + bia));
    }
  }
  __syncthreads();   // heads read rows across waves

  // ---------------- heads: per-row dot(256), scatter back to original rows ----------------
  if (tid < 128) {
    int hslot = glocal * 128 + tid;
    if (hslot < cnt) {
      int b = rowidx[bucketBase + hslot];
      const float* wb = hWb + c * H_W;
      const float* wsp = hWs + c * H_W;
      const float* wt = hWt + c * H_W;
      const u32x4* hr4 = (const u32x4*)(sH + tid * 264);   // tid*528 bytes, 16B aligned
      float d0 = 0.f, d1 = 0.f, d2 = 0.f;
#pragma unroll 4
      for (int jq = 0; jq < 32; ++jq) {
        u32x4 v = hr4[jq];
        unsigned w4[4] = {v.x, v.y, v.z, v.w};
#pragma unroll
        for (int e = 0; e < 4; ++e) {
          int j = jq * 8 + e * 2;
          float s0 = bf2f((unsigned short)(w4[e] & 0xFFFFu));
          float s1 = bf2f((unsigned short)(w4[e] >> 16));
          d0 += s0 * wb[j]  + s1 * wb[j + 1];
          d1 += s0 * wsp[j] + s1 * wsp[j + 1];
          d2 += s0 * wt[j]  + s1 * wt[j + 1];
        }
      }
      out[b]           = sigm(d0 + hbb[c]);
      out[B_N + b]     = tanhf(d1 + hbs[c]);
      out[2 * B_N + b] = sigm(d2 + hbt[c]);
    }
  }
}

// ---------------- workspace layout ----------------
static const size_t OFF_ROWIDX = 256;
static const size_t OFF_SA  = OFF_ROWIDX + (size_t)B_N * 4;
static const size_t OFF_SB  = OFF_SA  + (size_t)B_N * D_SPD * 2;
static const size_t OFF_WS2 = OFF_SB  + (size_t)B_N * D_SPD * 2;
static const size_t OFF_WS3 = OFF_WS2 + (size_t)D_SPD * D_SPD * 2;
static const size_t OFF_W1  = OFF_WS3 + (size_t)D_SPD * D_SPD * 2;
static const size_t OFF_W2  = OFF_W1  + (size_t)N_C * D_IN * H_W * 2;
static const size_t OFF_W3  = OFF_W2  + (size_t)N_C * H_W * H_W * 2;

extern "C" void kernel_launch(void* const* d_in, const int* in_sizes, int n_in,
                              void* d_out, int out_size, void* d_ws, size_t ws_size,
                              hipStream_t stream) {
  (void)in_sizes; (void)n_in; (void)out_size; (void)ws_size;
  const float* rgb    = (const float*)d_in[0];
  const float* spd    = (const float*)d_in[1];
  const int*   cmd    = (const int*)d_in[2];
  const float* spd_W1 = (const float*)d_in[3];
  const float* spd_b1 = (const float*)d_in[4];
  const float* spd_W2 = (const float*)d_in[5];
  const float* spd_b2 = (const float*)d_in[6];
  const float* spd_W3 = (const float*)d_in[7];
  const float* spd_b3 = (const float*)d_in[8];
  const float* W1  = (const float*)d_in[9];
  const float* b1  = (const float*)d_in[10];
  const float* W2  = (const float*)d_in[11];
  const float* b2  = (const float*)d_in[12];
  const float* W3  = (const float*)d_in[13];
  const float* b3  = (const float*)d_in[14];
  const float* hWb = (const float*)d_in[15];
  const float* hbb = (const float*)d_in[16];
  const float* hWs = (const float*)d_in[17];
  const float* hbs = (const float*)d_in[18];
  const float* hWt = (const float*)d_in[19];
  const float* hbt = (const float*)d_in[20];
  float* out = (float*)d_out;

  char* ws = (char*)d_ws;
  int* hdr = (int*)ws;
  int* rowidx = (int*)(ws + OFF_ROWIDX);
  unsigned short* sA    = (unsigned short*)(ws + OFF_SA);
  unsigned short* sBuf  = (unsigned short*)(ws + OFF_SB);
  unsigned short* ws2bf = (unsigned short*)(ws + OFF_WS2);
  unsigned short* ws3bf = (unsigned short*)(ws + OFF_WS3);
  unsigned short* w1bf  = (unsigned short*)(ws + OFF_W1);
  unsigned short* w2bf  = (unsigned short*)(ws + OFF_W2);
  unsigned short* w3bf  = (unsigned short*)(ws + OFF_W3);

  k_init<<<1, 32, 0, stream>>>(hdr);

  int n;
  n = D_SPD * D_SPD;    k_cvt<<<(n + 255) / 256, 256, 0, stream>>>(spd_W2, ws2bf, n);
  n = D_SPD * D_SPD;    k_cvt<<<(n + 255) / 256, 256, 0, stream>>>(spd_W3, ws3bf, n);
  n = N_C * D_IN * H_W; k_cvt<<<(n + 255) / 256, 256, 0, stream>>>(W1, w1bf, n);
  n = N_C * H_W * H_W;  k_cvt<<<(n + 255) / 256, 256, 0, stream>>>(W2, w2bf, n);
  n = N_C * H_W * H_W;  k_cvt<<<(n + 255) / 256, 256, 0, stream>>>(W3, w3bf, n);

  k_spd1<<<(B_N * D_SPD) / 256, 256, 0, stream>>>(spd, spd_W1, spd_b1, sA);
  k_hist<<<B_N / 256, 256, 0, stream>>>(cmd, hdr);
  k_scan<<<1, 1, 0, stream>>>(hdr);
  k_scatter<<<B_N / 256, 256, 0, stream>>>(cmd, hdr, rowidx);

  k_spd_gemm<<<B_N / 128, 256, 0, stream>>>(sA, ws2bf, spd_b2, sBuf);   // s1 -> s2
  k_spd_gemm<<<B_N / 128, 256, 0, stream>>>(sBuf, ws3bf, spd_b3, sA);   // s2 -> s3

  k_branch<<<MAXG, 256, 0, stream>>>(rgb, sA, hdr, rowidx, w1bf, w2bf, w3bf,
                                     b1, b2, b3, hWb, hbb, hWs, hbs, hWt, hbt, out);
}
